// Net_LV_75797582840291
// MI455X (gfx1250) — matrix operand order repaired and checked by reading, not compiled, not run on hardware
//
#include <hip/hip_runtime.h>

typedef __attribute__((ext_vector_type(16))) _Float16 v16h;
typedef __attribute__((ext_vector_type(8)))  _Float16 v8h;
typedef __attribute__((ext_vector_type(8)))  float    v8f;

#define MC_N    32768
#define NSTEPS  90
#define NGRID   91

// ---- output offsets (floats) in d_out, concatenated in reference return order
#define OFF_PATH   0
#define OFF_VAR    (MC_N*NGRID)
#define OFF_DLAST  (2*MC_N*NGRID)
#define OFF_PRICE  (OFF_DLAST + MC_N)
#define OFF_VARP   (OFF_PRICE + 63)
#define OFF_EXO    (OFF_VARP + 63)
#define OFF_MEAN   (OFF_EXO + MC_N)
#define OFF_VARTOT (OFF_MEAN + 1)
#define OFF_ERR    (OFF_VARTOT + 1)

// ---- packed-weight half offsets within one maturity block (26112 halves total)
#define WM_HALVES 26112
#define O_dWi 0        // 1x4 frags
#define O_dWh 2048     // 3 layers of 2x4 frags, + l*4096
#define O_dWo 14336    // 2x1
#define O_vWi 15360    // 1x2
#define O_vWh 16384    // + l*1024 (1x2 each)
#define O_vWo 18432    // 1x4
#define O_eWi 20480    // 3x2
#define O_eWh 23552    // + l*1024
#define O_eWo 25600    // 1x1

// ---- packed bias float offsets within one maturity block (544 floats total)
#define BM_FLOATS 544
#define B_dbi 0
#define B_dbh 64
#define B_dbo 256
#define B_vbi 272
#define B_vbh 304
#define B_vbo 368
#define B_ebi 432
#define B_ebh 464
#define B_ebo 528

// ---------------------------------------------------------------------------
// Weight packer: f32 [K x N] row-major  ->  f16 B-fragments (32x16 per frag).
// Fragment order: frag = nt*KT + kt.  Lane layout per CDNA5 ISA (B matrix):
// lane group g=lane/16 holds K=8g..8g+7 then K=16+8g..16+8g+7 of column n=lane%16.
__global__ void pack_weights_k(const float* __restrict__ src, _Float16* __restrict__ dst,
                               int K, int N, int KT, int NT) {
  int tid = blockIdx.x*blockDim.x + threadIdx.x;
  int frag = tid >> 5, lane = tid & 31;
  if (frag >= KT*NT) return;
  int nt = frag / KT, kt = frag % KT;
  int g = lane >> 4, nn = nt*16 + (lane & 15);
  union { v16h v; _Float16 h[16]; } u;
#pragma unroll
  for (int e = 0; e < 16; ++e) {
    int k = kt*32 + ((e < 8) ? (g*8 + e) : (16 + g*8 + (e - 8)));
    float w = (k < K && nn < N) ? src[k*N + nn] : 0.0f;
    u.h[e] = (_Float16)w;
  }
  *(v16h*)(dst + (size_t)frag*512 + lane*16) = u.v;
}

__global__ void pack_bias_k(const float* __restrict__ src, float* __restrict__ dst,
                            int N, int Npad) {
  int t = threadIdx.x;
  if (t < Npad) dst[t] = (t < N) ? src[t] : 0.0f;
}

__global__ void zero_accum_k(float* a) { int t = threadIdx.x; if (t < 130) a[t] = 0.0f; }

// ---------------------------------------------------------------------------
// A fragment (16x32 f16) from m-major LDS buffer, stride 96 halves.
// ISA layout: lane m%16, group g=lane/16; halves 0..7 = K (kt*32+g*8)..+7,
// halves 8..15 = K (kt*32+16+g*8)..+7  -> two contiguous ds_load_b128.
__device__ __forceinline__ v16h load_A_frag(const _Float16* in, int stride, int kt, int lane) {
  int m = lane & 15, g = lane >> 4;
  const _Float16* p = in + m*stride + kt*32 + g*8;
  union { v16h v; v8h h[2]; } u;
  u.h[0] = *(const v8h*)p;
  u.h[1] = *(const v8h*)(p + 16);
  return u.v;
}

__device__ __forceinline__ v16h load_B_frag(const _Float16* wf, int frag, int lane) {
  return *(const v16h*)(wf + frag*512 + (lane << 4));
}

// One dense layer: [16 x 32*KT] x [32*KT x 16*NT] + bias (+ReLU), in/out m-major
// stride-96 LDS buffers (aliasing safe: A is register-resident before stores).
template<int KT, int NT, bool RELU>
__device__ __forceinline__ void mlp_layer(const _Float16* in, const _Float16* wf,
                                          const float* bias, _Float16* out, int lane) {
  v16h A[KT];
#pragma unroll
  for (int kt = 0; kt < KT; ++kt) A[kt] = load_A_frag(in, 96, kt, lane);
  int n16 = lane & 15, g = lane >> 4;
#pragma unroll
  for (int nt = 0; nt < NT; ++nt) {
    v8f c = {};
#pragma unroll
    for (int kt = 0; kt < KT; ++kt)
      c = __builtin_amdgcn_wmma_f32_16x16x32_f16(false, A[kt], false,
            load_B_frag(wf, nt*KT + kt, lane), (short)0, c, false, false);
    float bs = bias[nt*16 + n16];
#pragma unroll
    for (int v = 0; v < 8; ++v) {
      float x = c[v] + bs;
      if (RELU) x = fmaxf(x, 0.0f);
      out[(v + 8*g)*96 + nt*16 + n16] = (_Float16)x;   // C layout: row m=v+8g, col n16
    }
  }
}

// ---------------------------------------------------------------------------
// Main Monte-Carlo kernel: 256 blocks x 256 threads; 8 waves/block, 16 paths/wave.
__global__ __launch_bounds__(256) void mc_kernel(
    const float* __restrict__ S0, const float* __restrict__ z,
    const _Float16* __restrict__ wpk, const float* __restrict__ bpk,
    float* __restrict__ accum, float* __restrict__ out) {
  extern __shared__ char smem[];
  _Float16* w_lds = (_Float16*)smem;                  // 26112 halves  (52224 B)
  float*    b_lds = (float*)(smem + 52224);           // 544 f         (2176 B)
  _Float16* pathb = (_Float16*)(smem + 54400);        // 128 x 96 f16  (24576 B)
  float*    cvbuf = (float*)(smem + 78976);           // 128 x 64 f32  (32768 B)
  _Float16* actb  = (_Float16*)(smem + 111744);       // 8 x 16 x 96   (24576 B)
  float*    pf    = (float*)(smem + 136320);          // 2688 f scratch(10752 B)
  float*    Sb    = (float*)(smem + 147072);          // 6 x 128 f32   (3072 B)
  float* rmaxb = Sb + 128; float* cveb = Sb + 256; float* diffv = Sb + 384;
  float* coeff = Sb + 512; float* dWb  = Sb + 640;    // total 150144 B

  int tid = threadIdx.x, lane = tid & 31, wv = tid >> 5;
  int pbase = blockIdx.x * 128;
  int lp = wv*16 + (lane & 15);       // block-local path (meaningful for lane<16)
  int gp = pbase + lp;                // global path
  _Float16* awave = actb  + wv*1536;  // this wave's activation tile (16 x 96)
  _Float16* prow  = pathb + wv*1536;  // this wave's path-feature rows

  float s0 = S0[0];
  for (int t = tid; t < 1536; t += 256) ((uint4*)pathb)[t] = make_uint4(0u,0u,0u,0u);
  for (int t = tid; t < 8192; t += 256) cvbuf[t] = 0.0f;
  if (tid < 128) { Sb[tid] = s0; rmaxb[tid] = s0; cveb[tid] = 0.0f; }
  __syncthreads();
  if (tid < 128) {
    pathb[tid*96 + 1] = (_Float16)s0;                  // feature 1 = path[:,0] = S0
    out[OFF_PATH + (pbase + tid)*91] = s0;
  }
  __syncthreads();

  for (int i = 0; i < NSTEPS; ++i) {
    int mm = i / 30;
    if ((i % 30) == 0) {                               // stage this period's weights
      __syncthreads();
      const uint4* ws4 = (const uint4*)(wpk + mm*WM_HALVES);
      for (int t = tid; t < 3264; t += 256) ((uint4*)w_lds)[t] = ws4[t];
      const float* bs = bpk + mm*BM_FLOATS;
      for (int t = tid; t < BM_FLOATS; t += 256) b_lds[t] = bs[t];
      __syncthreads();
    }
    float tc   = 0.01f * (float)i;                     // t_prev
    float disc = __expf(-0.025f * tc);

    // ---- per-path prep (lanes 0..15 of each wave)
    if (lane < 16) {
      float sv = Sb[lp];
      float zc = z[gp*90 + i];
      dWb[lp] = 0.1f * zc;                             // sqrt(h)=0.1
      if (i < 89) __builtin_prefetch(&z[gp*90 + i + 1], 0, 1);
      _Float16* row = awave + (lane & 15)*96;
      row[0] = (_Float16)tc;  row[1] = (_Float16)sv;
#pragma unroll
      for (int k2 = 2; k2 < 32; ++k2) row[k2] = (_Float16)0.0f;
      prow[(lane & 15)*96] = (_Float16)tc;             // exotic feature 0 = t
    }

    // ---- diffusion net (2->64, 3x 64->64, 64->16), softplus out
    mlp_layer<1,4,true>(awave, w_lds + O_dWi, b_lds + B_dbi, awave, lane);
#pragma unroll
    for (int l = 0; l < 3; ++l)
      mlp_layer<2,4,true>(awave, w_lds + O_dWh + l*4096, b_lds + B_dbh + l*64, awave, lane);
    {
      v16h a0 = load_A_frag(awave, 96, 0, lane);
      v16h a1 = load_A_frag(awave, 96, 1, lane);
      v8f c = {};
      c = __builtin_amdgcn_wmma_f32_16x16x32_f16(false, a0, false, load_B_frag(w_lds + O_dWo, 0, lane), (short)0, c, false, false);
      c = __builtin_amdgcn_wmma_f32_16x16x32_f16(false, a1, false, load_B_frag(w_lds + O_dWo, 1, lane), (short)0, c, false, false);
      float b0 = b_lds[B_dbo + (lane & 15)];
      if ((lane & 15) == 0) {
        int g = lane >> 4;
#pragma unroll
        for (int v = 0; v < 8; ++v) {
          float y = c[v] + b0;
          float sp = (y > 15.0f) ? y : __logf(1.0f + __expf(y));
          diffv[wv*16 + v + 8*g] = sp;
        }
      }
    }

    // ---- Euler step, coeff, var_path / path outputs
    if (lane < 16) {
      float dv = diffv[lp], sv = Sb[lp], dW = dWb[lp];
      float V = dv * dv;
      out[OFF_VAR + gp*91 + i + 1] = V;
      if (i == 0)  out[OFF_VAR + gp*91] = V;           // var0 == step-1 V
      if (i == 89) out[OFF_DLAST + gp] = dv;
      float sn = sv + 0.025f*sv*0.01f/(1.0f + 0.025f*sv*0.1f)
                    + sv*dv*dW/(1.0f + sv*dv*0.1f);
      coeff[lp] = disc * sv * dv * dW;
      Sb[lp] = sn;
      rmaxb[lp] = fmaxf(rmaxb[lp], sn);
      out[OFF_PATH + gp*91 + i + 1] = sn;
      _Float16* row = awave + (lane & 15)*96;          // rebuild (t, S_old) input
      row[0] = (_Float16)tc;  row[1] = (_Float16)sv;
#pragma unroll
      for (int k2 = 2; k2 < 32; ++k2) row[k2] = (_Float16)0.0f;
    }

    // ---- vanilla-CV net (2->32, 2x 32->32, 32->64), accumulate cv_v
    mlp_layer<1,2,true>(awave, w_lds + O_vWi, b_lds + B_vbi, awave, lane);
#pragma unroll
    for (int l = 0; l < 2; ++l)
      mlp_layer<1,2,true>(awave, w_lds + O_vWh + l*1024, b_lds + B_vbh + l*32, awave, lane);
    {
      v16h a0 = load_A_frag(awave, 96, 0, lane);
      int g = lane >> 4;
#pragma unroll
      for (int nt = 0; nt < 4; ++nt) {
        v8f c = {};
        c = __builtin_amdgcn_wmma_f32_16x16x32_f16(false, a0, false, load_B_frag(w_lds + O_vWo, nt, lane), (short)0, c, false, false);
        int n = nt*16 + (lane & 15);
        float bsv = b_lds[B_vbo + n];
        if (n < 63) {
#pragma unroll
          for (int v = 0; v < 8; ++v) {
            int lq = wv*16 + v + 8*g;
            cvbuf[lq*64 + n] += coeff[lq] * (c[v] + bsv);
          }
        }
      }
    }

    // ---- exotic-CV net (96->32 from path features, 2x 32->32, 32->16)
    mlp_layer<3,2,true>(prow, w_lds + O_eWi, b_lds + B_ebi, awave, lane);
#pragma unroll
    for (int l = 0; l < 2; ++l)
      mlp_layer<1,2,true>(awave, w_lds + O_eWh + l*1024, b_lds + B_ebh + l*32, awave, lane);
    {
      v16h a0 = load_A_frag(awave, 96, 0, lane);
      v8f c = {};
      c = __builtin_amdgcn_wmma_f32_16x16x32_f16(false, a0, false, load_B_frag(w_lds + O_eWo, 0, lane), (short)0, c, false, false);
      float b0 = b_lds[B_ebo + (lane & 15)];
      if ((lane & 15) == 0) {
        int g = lane >> 4;
#pragma unroll
        for (int v = 0; v < 8; ++v) {
          int lq = wv*16 + v + 8*g;
          cveb[lq] += coeff[lq] * (c[v] + b0);
        }
      }
    }

    // path[:, i+1] = S_new written AFTER cv_e (matches reference ordering)
    if (lane < 16) prow[(lane & 15)*96 + i + 2] = (_Float16)Sb[lp];

    // ---- period end: vanilla prices, block-reduce, atomics
    if ((i % 30) == 29) {
      if (lane < 16) {
        float sv = Sb[lp];
        float dm = __expf(-0.025f * 0.01f * (float)(i + 1));
#pragma unroll
        for (int k = 0; k < 21; ++k) {
          float st = 0.8f + 0.02f*(float)k;
          pf[lp*21 + k] = dm * fmaxf(sv - st, 0.0f) - cvbuf[lp*64 + mm*21 + k];
        }
      }
      __syncthreads();
      if (tid < 21) {
        float s = 0.0f, sq = 0.0f;
        for (int p = 0; p < 128; ++p) { float v = pf[p*21 + tid]; s += v; sq += v*v; }
        atomicAdd(&accum[mm*21 + tid], s);
        atomicAdd(&accum[63 + mm*21 + tid], sq);
      }
      __syncthreads();
    }
  }

  // ---- exotic payoff + block reductions for mean/var/error
  if (lane < 16) {
    float discT = __expf(-0.025f * 0.9f);
    float ep = discT * (rmaxb[lp] - Sb[lp]) - cveb[lp];
    out[OFF_EXO + gp] = ep;
    pf[lp] = ep; pf[128 + lp] = ep*ep; pf[256 + lp] = cveb[lp];
  }
  __syncthreads();
  if (tid < 3) {
    float s = 0.0f;
    for (int p = 0; p < 128; ++p) s += pf[tid*128 + p];
    atomicAdd(&accum[126 + tid], s);
  }
}

// ---------------------------------------------------------------------------
__global__ void finalize_k(const float* __restrict__ accum, float* __restrict__ out,
                           float* __restrict__ meanraw) {
  int t = threadIdx.x;
  const float n = 32768.0f, n1 = 32767.0f;
  if (t < 63) {
    float s = accum[t], sq = accum[63 + t];
    float mean = s / n;
    out[OFF_PRICE + t] = mean;
    out[OFF_VARP  + t] = (sq - s*mean) / n1;           // ddof=1
  } else if (t == 63) {
    float s = accum[126], sq = accum[127], sc = accum[128];
    float mean = s / n;
    out[OFF_MEAN]   = mean;
    out[OFF_VARTOT] = (sq - s*mean) / n1;
    meanraw[0] = (s + sc) / n;                         // mean(discT*exotic)
  }
}

// error = discT*exotic - mean(discT*exotic) - cv_e  ==  exotic_price - mean_raw
__global__ void error_k(const float* __restrict__ meanraw, float* __restrict__ out) {
  int p = blockIdx.x*blockDim.x + threadIdx.x;
  if (p < MC_N) out[OFF_ERR + p] = out[OFF_EXO + p] - meanraw[0];
}

// ---------------------------------------------------------------------------
extern "C" void kernel_launch(void* const* d_in, const int* in_sizes, int n_in,
                              void* d_out, int out_size, void* d_ws, size_t ws_size,
                              hipStream_t stream) {
  const float* S0  = (const float*)d_in[0];
  const float* z   = (const float*)d_in[1];
  const float* dWi = (const float*)d_in[5];
  const float* dbi = (const float*)d_in[6];
  const float* dWh = (const float*)d_in[7];
  const float* dbh = (const float*)d_in[8];
  const float* dWo = (const float*)d_in[9];
  const float* dbo = (const float*)d_in[10];
  const float* vWi = (const float*)d_in[11];
  const float* vbi = (const float*)d_in[12];
  const float* vWh = (const float*)d_in[13];
  const float* vbh = (const float*)d_in[14];
  const float* vWo = (const float*)d_in[15];
  const float* vbo = (const float*)d_in[16];
  const float* eWi = (const float*)d_in[17];
  const float* ebi = (const float*)d_in[18];
  const float* eWh = (const float*)d_in[19];
  const float* ebh = (const float*)d_in[20];
  const float* eWo = (const float*)d_in[21];
  const float* ebo = (const float*)d_in[22];
  float* out = (float*)d_out;

  char* ws = (char*)d_ws;
  _Float16* wpk  = (_Float16*)ws;                      // 3*26112 halves = 156672 B
  float*    bpk  = (float*)(ws + 156672);              // 3*544 floats   = 6528 B
  float*   accum = (float*)(ws + 156672 + 6528);       // 130 floats
  float* meanraw = accum + 129;

  zero_accum_k<<<1, 256, 0, stream>>>(accum);

  for (int m = 0; m < 3; ++m) {
    _Float16* w = wpk + m*WM_HALVES;
    float*    b = bpk + m*BM_FLOATS;
    auto PW = [&](const float* src, int K, int N, int KT, int NT, int off) {
      int nthreads = KT*NT*32;
      pack_weights_k<<<(nthreads + 255)/256, 256, 0, stream>>>(src, w + off, K, N, KT, NT);
    };
    PW(dWi + m*2*50,               2, 50, 1, 4, O_dWi);
    for (int l = 0; l < 3; ++l) PW(dWh + (m*3 + l)*2500, 50, 50, 2, 4, O_dWh + l*4096);
    PW(dWo + m*50,                50,  1, 2, 1, O_dWo);
    PW(vWi + m*2*30,               2, 30, 1, 2, O_vWi);
    for (int l = 0; l < 2; ++l) PW(vWh + (m*2 + l)*900,  30, 30, 1, 2, O_vWh + l*1024);
    PW(vWo + m*30*63,             30, 63, 1, 4, O_vWo);
    PW(eWi + m*92*20,             92, 20, 3, 2, O_eWi);
    for (int l = 0; l < 2; ++l) PW(eWh + (m*2 + l)*400,  20, 20, 1, 2, O_eWh + l*1024);
    PW(eWo + m*20,                20,  1, 1, 1, O_eWo);
    auto PB = [&](const float* src, int N, int Np, int off) {
      pack_bias_k<<<1, 128, 0, stream>>>(src, b + off, N, Np);
    };
    PB(dbi + m*50, 50, 64, B_dbi);
    for (int l = 0; l < 3; ++l) PB(dbh + (m*3 + l)*50, 50, 64, B_dbh + l*64);
    PB(dbo + m, 1, 16, B_dbo);
    PB(vbi + m*30, 30, 32, B_vbi);
    for (int l = 0; l < 2; ++l) PB(vbh + (m*2 + l)*30, 30, 32, B_vbh + l*32);
    PB(vbo + m*63, 63, 64, B_vbo);
    PB(ebi + m*20, 20, 32, B_ebi);
    for (int l = 0; l < 2; ++l) PB(ebh + (m*2 + l)*20, 20, 32, B_ebh + l*32);
    PB(ebo + m, 1, 16, B_ebo);
  }

  constexpr size_t SMEM = 150144;
  hipFuncSetAttribute((const void*)mc_kernel,
                      hipFuncAttributeMaxDynamicSharedMemorySize, (int)SMEM);
  mc_kernel<<<256, 256, SMEM, stream>>>(S0, z, wpk, bpk, accum, out);
  finalize_k<<<1, 64, 0, stream>>>(accum, out, meanraw);
  error_k<<<MC_N/256, 256, 0, stream>>>(meanraw, out);
  (void)in_sizes; (void)n_in; (void)out_size; (void)ws_size;
}
